// StructureEncoder_62474594287670
// MI455X (gfx1250) — matrix operand order repaired and checked by reading, not compiled, not run on hardware
//
#include <hip/hip_runtime.h>

typedef __attribute__((ext_vector_type(16))) _Float16 v16h;
typedef __attribute__((ext_vector_type(8)))  float    v8f;
typedef __attribute__((ext_vector_type(2)))  _Float16 h2;
typedef __attribute__((ext_vector_type(8)))  unsigned u32x8;

#define NJ 21
#define WS_PER_JOINT 4096            // A1(1KB) | A2(1KB) | C1(1KB) | C2(1KB)
#define WS_NEEDED   (NJ * WS_PER_JOINT)

static constexpr int PARENTS_H[NJ] =
    {-1, 0, 0, 0, 1, 2, 3, 4, 5, 6, 7, 8, 9, 9, 9, 12, 13, 14, 16, 17, 18};

__device__ __forceinline__ v16h vzero16() {
    v16h v;
#pragma unroll
    for (int i = 0; i < 16; ++i) v[i] = (_Float16)0.0f;
    return v;
}

__device__ __forceinline__ float relu1(float x) {
    return __builtin_amdgcn_fmed3f(x, 0.0f, 3.0e38f);   // single v_med3
}

__device__ __forceinline__ unsigned pack2(float a, float b) {
    h2 p; p.x = (_Float16)a; p.y = (_Float16)b;          // -> v_cvt_pk_f16_f32
    return __builtin_bit_cast(unsigned, p);
}

__device__ __forceinline__ unsigned relu_pk(unsigned v) { // packed-f16 relu
#if __has_builtin(__builtin_elementwise_max)
    h2 p = __builtin_bit_cast(h2, v);
    h2 z; z.x = (_Float16)0.0f; z.y = (_Float16)0.0f;
    h2 r = __builtin_elementwise_max(p, z);               // -> v_pk_max_num_f16
    return __builtin_bit_cast(unsigned, r);
#else
    h2 p = __builtin_bit_cast(h2, v);
    h2 r;
    r.x = (p.x > (_Float16)0.0f) ? p.x : (_Float16)0.0f;
    r.y = (p.y > (_Float16)0.0f) ? p.y : (_Float16)0.0f;
    return __builtin_bit_cast(unsigned, r);
#endif
}

// ---------------------------------------------------------------------------
// Prep: format WMMA operand images once per launch. Grid: <<<NJ, 32>>>.
// ---------------------------------------------------------------------------
__global__ void se_prep_kernel(const float* __restrict__ W1_root,
                               const float* __restrict__ b1_root,
                               const float* __restrict__ W1,
                               const float* __restrict__ b1,
                               const float* __restrict__ W2,
                               const float* __restrict__ b2,
                               char* __restrict__ ws)
{
    const int  i  = blockIdx.x;      // joint
    const int  L  = threadIdx.x;     // lane 0..31
    const bool lo = L < 16;
    const int  m  = L & 15;

    v16h A1 = vzero16();
    if (i == 0) {
        if (lo && m < 10) {
#pragma unroll
            for (int k = 0; k < 4; ++k) A1[k] = (_Float16)W1_root[m * 4 + k];
        }
    } else {
        const float* w = W1 + (i - 1) * 100;
        if (m < 10) {
            if (lo) {
#pragma unroll
                for (int k = 0; k < 8; ++k) A1[k] = (_Float16)w[m * 10 + k];
                A1[8] = (_Float16)w[m * 10 + 8];
                A1[9] = (_Float16)w[m * 10 + 9];
            }
        }
    }

    v16h A2 = vzero16();
    {
        const float* w = W2 + i * 60;
        if (m < 6) {
            if (lo) {
#pragma unroll
                for (int k = 0; k < 8; ++k) A2[k] = (_Float16)w[m * 10 + k];
                A2[8] = (_Float16)w[m * 10 + 8];
                A2[9] = (_Float16)w[m * 10 + 9];
            }
        }
    }

    v8f C1;
    {
        const float* bp = (i == 0) ? b1_root : (b1 + (i - 1) * 10);
#pragma unroll
        for (int r = 0; r < 8; ++r) {
            const int idx = r + (lo ? 0 : 8);
            float bv = 0.0f;
            if (idx < 10) bv = bp[idx];
            C1[r] = bv;
        }
    }
    v8f C2;
#pragma unroll
    for (int r = 0; r < 8; ++r) {
        float bv = 0.0f;
        if (r < 6) { if (lo) bv = b2[i * 6 + r]; }
        C2[r] = bv;
    }

    char* base = ws + i * WS_PER_JOINT + L * 32;
    *reinterpret_cast<v16h*>(base)        = A1;
    *reinterpret_cast<v16h*>(base + 1024) = A2;
    *reinterpret_cast<v8f*>(base + 2048)  = C1;
    *reinterpret_cast<v8f*>(base + 3072)  = C2;
}

// ---------------------------------------------------------------------------
// Phase 1: build B1 = [quat | parent feat] and run layer-1 WMMA.
// ---------------------------------------------------------------------------
__device__ __forceinline__ v8f phase_l1(
    int i, int p, bool hasParent,
    const v16h& A1, const v8f& C1,
    const float* __restrict__ qp,       // quat + row*84
    const unsigned (*fp)[3],            // packed feats, pre-masked
    unsigned mask)
{
    const float4 q = *reinterpret_cast<const float4*>(qp + i * 4);
    u32x8 b1;
    b1[0] = pack2(q.x, q.y) & mask;
    b1[1] = pack2(q.z, q.w) & mask;
    if (hasParent) { b1[2] = fp[p][0]; b1[3] = fp[p][1]; b1[4] = fp[p][2]; }
    else           { b1[2] = 0u;       b1[3] = 0u;       b1[4] = 0u;       }
    b1[5] = 0u; b1[6] = 0u; b1[7] = 0u;
    return __builtin_amdgcn_wmma_f32_16x16x32_f16(
        false, A1, false, __builtin_bit_cast(v16h, b1), (short)0, C1,
        false, false);
}

// ---------------------------------------------------------------------------
// Phase 2: D1 -> B2 (f16 relu; rows 8,9 shipped from lanes 16-31 via bpermute).
// ---------------------------------------------------------------------------
__device__ __forceinline__ v16h phase_mid(const v8f& D1, unsigned mask, int lane)
{
    const unsigned p0 = relu_pk(pack2(D1[0], D1[1]));
    const unsigned p1 = relu_pk(pack2(D1[2], D1[3]));
    const unsigned p2 = relu_pk(pack2(D1[4], D1[5]));
    const unsigned p3 = relu_pk(pack2(D1[6], D1[7]));
    const unsigned h89 = (unsigned)__builtin_amdgcn_ds_bpermute(
        (lane | 16) << 2, (int)p0);
    u32x8 b2;
    b2[0] = p0 & mask; b2[1] = p1 & mask; b2[2] = p2 & mask; b2[3] = p3 & mask;
    b2[4] = h89 & mask;
    b2[5] = 0u; b2[6] = 0u; b2[7] = 0u;
    return __builtin_bit_cast(v16h, b2);
}

// ---------------------------------------------------------------------------
// Phase 3: relu D2, pack feats for children, store 6 outputs.
// ---------------------------------------------------------------------------
__device__ __forceinline__ void phase_fin(
    int i, const v8f& D2, float* __restrict__ op,
    unsigned (*fp)[3], unsigned mask, bool valid)
{
    const float d0 = relu1(D2[0]); const float d1 = relu1(D2[1]);
    const float d2 = relu1(D2[2]); const float d3 = relu1(D2[3]);
    const float d4 = relu1(D2[4]); const float d5 = relu1(D2[5]);

    fp[i][0] = pack2(d0, d1) & mask;
    fp[i][1] = pack2(d2, d3) & mask;
    fp[i][2] = pack2(d4, d5) & mask;

    if (valid) {
        float* o = op + i * 6;
        float2 s0; s0.x = d0; s0.y = d1;
        float2 s1; s1.x = d2; s1.y = d3;
        float2 s2; s2.x = d4; s2.y = d5;
        *reinterpret_cast<float2*>(o)     = s0;
        *reinterpret_cast<float2*>(o + 2) = s1;
        *reinterpret_cast<float2*>(o + 4) = s2;
    }
}

// ---------------------------------------------------------------------------
// Main: one wave == 2 independent 16-row tiles, interleaved so each WMMA's
// RAW-hazard shadow is filled by the other tile's work. Weight operands are
// loaded once per joint and reused by both tiles.
// ---------------------------------------------------------------------------
__global__ __launch_bounds__(256) void se_wmma_main(
    const float* __restrict__ quat,   // [B,21,4]
    const char*  __restrict__ ws,     // formatted operands
    float* __restrict__ out,          // [B,126]
    int Btot)
{
    const int lane = threadIdx.x & 31;
    const int wave = blockIdx.x * (blockDim.x >> 5) + (threadIdx.x >> 5);
    const int row0 = wave << 5;               // 32 rows per wave
    if (row0 >= Btot) return;                 // wave-uniform
    const bool lo  = lane < 16;
    const int  col = lane & 15;
    const unsigned mask = lo ? 0xFFFFFFFFu : 0u;

    const int r0 = row0 + col;
    const int r1 = row0 + 16 + col;
    const bool v0 = lo && (r0 < Btot);
    const bool v1 = lo && (r1 < Btot);
    const int r0c = (r0 < Btot) ? r0 : (Btot - 1);
    const int r1c = (r1 < Btot) ? r1 : (Btot - 1);

    const float* qp0 = quat + (size_t)r0c * (NJ * 4);
    const float* qp1 = quat + (size_t)r1c * (NJ * 4);
    float* op0 = out + (size_t)r0c * 126;
    float* op1 = out + (size_t)r1c * 126;

    const char* wb = ws + lane * 32;

    unsigned fp0[NJ][3];
    unsigned fp1[NJ][3];

#pragma unroll
    for (int i = 0; i < NJ; ++i) {
        const int  p  = (PARENTS_H[i] < 0) ? 0 : PARENTS_H[i];
        const bool hp = (i > 0);

        const v16h A1 = *reinterpret_cast<const v16h*>(wb + i * WS_PER_JOINT);
        const v16h A2 = *reinterpret_cast<const v16h*>(wb + i * WS_PER_JOINT + 1024);
        const v8f  C1 = *reinterpret_cast<const v8f*>(wb + i * WS_PER_JOINT + 2048);
        const v8f  C2 = *reinterpret_cast<const v8f*>(wb + i * WS_PER_JOINT + 3072);

        // interleaved dual-tile pipeline
        const v8f D1a = phase_l1(i, p, hp, A1, C1, qp0, fp0, mask);
        const v8f D1b = phase_l1(i, p, hp, A1, C1, qp1, fp1, mask);

        const v16h B2a = phase_mid(D1a, mask, lane);
        const v16h B2b = phase_mid(D1b, mask, lane);

        const v8f D2a = __builtin_amdgcn_wmma_f32_16x16x32_f16(
            false, A2, false, B2a, (short)0, C2, false, false);
        const v8f D2b = __builtin_amdgcn_wmma_f32_16x16x32_f16(
            false, A2, false, B2b, (short)0, C2, false, false);

        phase_fin(i, D2a, op0, fp0, mask, v0);
        phase_fin(i, D2b, op1, fp1, mask, v1);
    }
}

// ---------------------------------------------------------------------------
// Fallback (ws too small): self-formatting, single tile per wave.
// ---------------------------------------------------------------------------
__global__ __launch_bounds__(256) void se_wmma_fallback(
    const float* __restrict__ quat, const float* __restrict__ W1_root,
    const float* __restrict__ b1_root, const float* __restrict__ W1,
    const float* __restrict__ b1, const float* __restrict__ W2,
    const float* __restrict__ b2, float* __restrict__ out, int Btot)
{
    const int lane = threadIdx.x & 31;
    const int wave = blockIdx.x * (blockDim.x >> 5) + (threadIdx.x >> 5);
    const int row0 = wave << 4;
    if (row0 >= Btot) return;
    const bool lo  = lane < 16;
    const int  col = lane & 15;
    const int  row = row0 + col;
    const bool rvalid = lo && (row < Btot);
    h2 fp[NJ][3];

#pragma unroll
    for (int i = 0; i < NJ; ++i) {
        const int p = (PARENTS_H[i] < 0) ? 0 : PARENTS_H[i];
        v16h B1 = vzero16();
        if (rvalid) {
            const float4 q = *reinterpret_cast<const float4*>(
                quat + (size_t)row * (NJ * 4) + i * 4);
            B1[0] = (_Float16)q.x; B1[1] = (_Float16)q.y;
            B1[2] = (_Float16)q.z; B1[3] = (_Float16)q.w;
            if (i > 0) {
                B1[4] = fp[p][0].x; B1[5] = fp[p][0].y;
                B1[6] = fp[p][1].x; B1[7] = fp[p][1].y;
                B1[8] = fp[p][2].x; B1[9] = fp[p][2].y;
            }
        }
        v16h A1 = vzero16();
        if (i == 0) {
            if (lo && col < 10) {
#pragma unroll
                for (int k = 0; k < 4; ++k) A1[k] = (_Float16)W1_root[col * 4 + k];
            }
        } else {
            const float* w = W1 + (i - 1) * 100;
            if (col < 10) {
                if (lo) {
#pragma unroll
                    for (int k = 0; k < 8; ++k) A1[k] = (_Float16)w[col * 10 + k];
                    A1[8] = (_Float16)w[col * 10 + 8];
                    A1[9] = (_Float16)w[col * 10 + 9];
                }
            }
        }
        v8f C1;
        {
            const float* bp = (i == 0) ? b1_root : (b1 + (i - 1) * 10);
#pragma unroll
            for (int r = 0; r < 8; ++r) {
                const int idx = r + (lo ? 0 : 8);
                float bv = 0.0f;
                if (idx < 10) bv = bp[idx];
                C1[r] = bv;
            }
        }
        v8f D1 = __builtin_amdgcn_wmma_f32_16x16x32_f16(
            false, A1, false, B1, (short)0, C1, false, false);
#pragma unroll
        for (int r = 0; r < 8; ++r) D1[r] = relu1(D1[r]);

        h2 p01; p01.x = (_Float16)D1[0]; p01.y = (_Float16)D1[1];
        const int got = __builtin_amdgcn_ds_bpermute(
            (lane | 16) << 2, __builtin_bit_cast(int, p01));
        const h2 h89 = __builtin_bit_cast(h2, got);

        v16h B2 = vzero16();
        if (lo) {
#pragma unroll
            for (int r = 0; r < 8; ++r) B2[r] = (_Float16)D1[r];
            B2[8] = h89.x; B2[9] = h89.y;
        }
        v16h A2 = vzero16();
        {
            const float* w = W2 + i * 60;
            if (col < 6) {
                if (lo) {
#pragma unroll
                    for (int k = 0; k < 8; ++k) A2[k] = (_Float16)w[col * 10 + k];
                    A2[8] = (_Float16)w[col * 10 + 8];
                    A2[9] = (_Float16)w[col * 10 + 9];
                }
            }
        }
        v8f C2;
#pragma unroll
        for (int r = 0; r < 8; ++r) {
            float bv = 0.0f;
            if (r < 6) { if (lo) bv = b2[i * 6 + r]; }
            C2[r] = bv;
        }
        v8f D2 = __builtin_amdgcn_wmma_f32_16x16x32_f16(
            false, A2, false, B2, (short)0, C2, false, false);
#pragma unroll
        for (int r = 0; r < 6; ++r) D2[r] = relu1(D2[r]);

        fp[i][0].x = (_Float16)D2[0]; fp[i][0].y = (_Float16)D2[1];
        fp[i][1].x = (_Float16)D2[2]; fp[i][1].y = (_Float16)D2[3];
        fp[i][2].x = (_Float16)D2[4]; fp[i][2].y = (_Float16)D2[5];

        if (rvalid) {
            float* o = out + (size_t)row * 126 + i * 6;
            float2 s0; s0.x = D2[0]; s0.y = D2[1];
            float2 s1; s1.x = D2[2]; s1.y = D2[3];
            float2 s2; s2.x = D2[4]; s2.y = D2[5];
            *reinterpret_cast<float2*>(o)     = s0;
            *reinterpret_cast<float2*>(o + 2) = s1;
            *reinterpret_cast<float2*>(o + 4) = s2;
        }
    }
}

extern "C" void kernel_launch(void* const* d_in, const int* in_sizes, int n_in,
                              void* d_out, int out_size, void* d_ws, size_t ws_size,
                              hipStream_t stream) {
    const float* quat    = (const float*)d_in[0];
    const float* W1_root = (const float*)d_in[1];
    const float* b1_root = (const float*)d_in[2];
    const float* W1      = (const float*)d_in[3];
    const float* b1      = (const float*)d_in[4];
    const float* W2      = (const float*)d_in[5];
    const float* b2      = (const float*)d_in[6];
    float* out           = (float*)d_out;

    const int Btot = in_sizes[0] / (NJ * 4);        // 524288

    if (ws_size >= (size_t)WS_NEEDED) {
        const int waves  = (Btot + 31) / 32;        // 32 rows per wave
        const int blocks = (waves + 7) / 8;         // 8 waves per 256-thread block
        se_prep_kernel<<<NJ, 32, 0, stream>>>(
            W1_root, b1_root, W1, b1, W2, b2, (char*)d_ws);
        se_wmma_main<<<blocks, 256, 0, stream>>>(
            quat, (const char*)d_ws, out, Btot);
    } else {
        const int waves  = (Btot + 15) / 16;
        const int blocks = (waves + 7) / 8;
        se_wmma_fallback<<<blocks, 256, 0, stream>>>(
            quat, W1_root, b1_root, W1, b1, W2, b2, out, Btot);
    }
}
